// PanguModel_64235530879377
// MI455X (gfx1250) — matrix operand order repaired and checked by reading, not compiled, not run on hardware
//
#include <hip/hip_runtime.h>
#include <hip/hip_bf16.h>

// Fused Pangu window-attention for MI455X (gfx1250, wave32, WMMA).
// One workgroup per (lon, window_type): 96*10 = 960 blocks, 256 threads (8 waves).
// All intermediates (qkv, scores, attn) live in 304KB of LDS (320KB/WGP on CDNA5).
// GEMMs/attention run on v_wmma_f32_16x16x32_bf16 with f32 accumulation.
// b1/b2 are staged with global_load_async_to_lds_b128 (ASYNCcnt DMA path);
// W1 panels are prefetched (global_prefetch_b8) one panel ahead of the WMMA loop.

typedef __attribute__((ext_vector_type(16))) __bf16 v16bf;
typedef __attribute__((ext_vector_type(8)))  __bf16 v8bf;
typedef __attribute__((ext_vector_type(8)))  float  v8f;

#define TT 144     // tokens per window (2*6*12)
#define DD 192     // model dim
#define HH 6       // heads
#define HD 32      // head dim
#define TP 160     // tokens padded to multiple of 32 (K-dim of attn@v)
#define NWIN 10
#define NLON 96

// A-fragment (16x32 bf16): per-lane 16 contiguous elems at k = 2c..2c+15,
// c = 8*(lane>=16). rp = pointer to this lane's row, kbase already added.
__device__ __forceinline__ v16bf load_a_frag(const __bf16* rp, int c) {
    v8bf lo = *reinterpret_cast<const v8bf*>(rp + 2 * c);
    v8bf hi = *reinterpret_cast<const v8bf*>(rp + 2 * c + 8);
    return __builtin_shufflevector(lo, hi, 0,1,2,3,4,5,6,7,8,9,10,11,12,13,14,15);
}

__global__ __launch_bounds__(256)
void pangu_fused(const float* __restrict__ x, const float* __restrict__ mask,
                 const float* __restrict__ W1, const float* __restrict__ b1,
                 const float* __restrict__ W2, const float* __restrict__ b2,
                 const float* __restrict__ btab, float* __restrict__ out)
{
    // Phase-overlaid scratch: GEMM1 (x-tile + W1 panel) == attention (scores f32 +
    // attn bf16) == GEMM2 (W2^T). g1 and a are 129,024 bytes each; g2 is 73,728.
    union Scratch {
        struct { __bf16 sx[TT][DD]; __bf16 wp[DD][DD]; } g1;   // 55296 + 73728
        struct { float  sc[TT][TT]; __bf16 at[TT][TP]; } a;    // 82944 + 46080
        struct { __bf16 w2[DD][DD];                    } g2;   // 73728
    };
    __shared__ alignas(32) Scratch u;
    __shared__ alignas(32) __bf16 qo[HH][TT][HD];  // q, later attention output (55296 B)
    __shared__ alignas(32) __bf16 kb[HH][TT][HD];  // k                          (55296 B)
    __shared__ alignas(32) __bf16 vt[HH][HD][TP];  // v transposed, token-padded (61440 B)
    __shared__ alignas(16) float b1s[3 * DD];
    __shared__ alignas(16) float b2s[DD];

    const int tid  = threadIdx.x;
    const int lane = tid & 31;
    // Uniform wave id: readfirstlane makes the tile loops provably wave-uniform so
    // EXEC stays all-ones around every WMMA (ISA requirement) and loops use scalar
    // branches instead of exec masking.
    const int wv   = __builtin_amdgcn_readfirstlane(tid >> 5);   // wave id 0..7
    const int lw   = blockIdx.x;         // l*10 + w
    const int l    = lw / NWIN;
    const int w    = lw % NWIN;
    const int l16  = lane & 15;          // N / M column within tile
    const int hi8  = (lane >> 4) << 3;   // 0 or 8 : C/D row base, A-chunk base
    const int kh16 = (lane >> 4) << 4;   // 0 or 16: B-fragment K-half
    const float scale = 0.17677669529663687f;  // 32^-0.5

    // ---- stage inputs -------------------------------------------------------
    // b1/b2 go to LDS unconverted -> use the CDNA5 async global->LDS DMA path.
    if (tid < (3 * DD) / 4) {
        unsigned long long ga = (unsigned long long)(uintptr_t)(b1 + 4 * tid);
        unsigned int       la = (unsigned int)(uintptr_t)&b1s[4 * tid];
        asm volatile("global_load_async_to_lds_b128 %0, %1, off"
                     :: "v"(la), "v"(ga) : "memory");
    }
    if (tid < DD / 4) {
        unsigned long long ga = (unsigned long long)(uintptr_t)(b2 + 4 * tid);
        unsigned int       la = (unsigned int)(uintptr_t)&b2s[4 * tid];
        asm volatile("global_load_async_to_lds_b128 %0, %1, off"
                     :: "v"(la), "v"(ga) : "memory");
    }
    const float* xg = x + (size_t)lw * TT * DD;
    {
        __bf16* sxf = &u.g1.sx[0][0];
        for (int i = tid; i < TT * DD; i += 256) sxf[i] = (__bf16)xg[i];
        // zero the token-padding of v^T (K padding of attn@v contributes 0)
        for (int i = tid; i < HH * HD * (TP - TT); i += 256) {
            int h = i / (HD * (TP - TT));
            int r = i % (HD * (TP - TT));
            vt[h][r / (TP - TT)][TT + r % (TP - TT)] = (__bf16)0.0f;
        }
    }
    asm volatile("s_wait_asynccnt 0x0" ::: "memory");   // b1s/b2s landed in LDS

    // ---- GEMM1: qkv = x @ W1 + b1, one 192-column panel per p (p=0:q 1:k 2:v)
    for (int p = 0; p < 3; ++p) {
        __syncthreads();
        for (int i = tid; i < DD * DD; i += 256) {       // W1 panel, transposed to [n][k]
            int n = i % DD, k = i / DD;
            u.g1.wp[n][k] = (__bf16)W1[(size_t)k * (3 * DD) + p * DD + n];
        }
        __syncthreads();
        if (p < 2) {                                     // prefetch next panel into L2/L0
            for (int i = tid; i < DD * 6; i += 256) {    // 6 x 128B lines per 192-f32 row
                int k = i / 6, c = (i % 6) * 32;
                __builtin_prefetch(&W1[(size_t)k * (3 * DD) + (p + 1) * DD + c], 0, 1);
            }
        }
        for (int t = wv; t < 9 * 12; t += 8) {           // 144/16 x 192/16 tiles
            int mt = t / 12, nt = t % 12;
            const __bf16* arow = &u.g1.sx[mt * 16 + l16][0];
            const __bf16* bcol = &u.g1.wp[nt * 16 + l16][0];
            v8f acc = {};
#pragma unroll
            for (int ks = 0; ks < 6; ++ks) {             // K = 192 = 6*32
                v16bf A = load_a_frag(arow + ks * 32, hi8);
                v16bf B = *reinterpret_cast<const v16bf*>(bcol + ks * 32 + kh16);
                acc = __builtin_amdgcn_wmma_f32_16x16x32_bf16(
                        false, A, false, B, (short)0, acc, false, false);
            }
            int coln = nt * 16 + l16;
            float bv = b1s[p * DD + coln];
            int head = nt >> 1;
            int dloc = (nt & 1) * 16 + l16;
            int mrow = mt * 16 + hi8;
            if (p == 0) {
#pragma unroll
                for (int r = 0; r < 8; ++r)
                    qo[head][mrow + r][dloc] = (__bf16)((acc[r] + bv) * scale);
            } else if (p == 1) {
#pragma unroll
                for (int r = 0; r < 8; ++r)
                    kb[head][mrow + r][dloc] = (__bf16)(acc[r] + bv);
            } else {
#pragma unroll
                for (int r = 0; r < 8; ++r)
                    vt[head][dloc][mrow + r] = (__bf16)(acc[r] + bv);
            }
        }
    }
    __syncthreads();

    // ---- attention, head-sequential ----------------------------------------
    const float* ml = mask + (size_t)l * TT * TT;
    for (int h = 0; h < HH; ++h) {
        // scores = (q*scale) @ k^T : 9x9 tiles, single K-step (HD = 32)
        for (int t = wv; t < 81; t += 8) {
            int mt = t / 9, nt = t % 9;
            v16bf A = load_a_frag(&qo[h][mt * 16 + l16][0], hi8);
            v16bf B = *reinterpret_cast<const v16bf*>(&kb[h][nt * 16 + l16][kh16]);
            v8f acc = {};
            acc = __builtin_amdgcn_wmma_f32_16x16x32_bf16(
                    false, A, false, B, (short)0, acc, false, false);
            int n = nt * 16 + l16, mb = mt * 16 + hi8;
#pragma unroll
            for (int r = 0; r < 8; ++r) u.a.sc[mb + r][n] = acc[r];
        }
        __syncthreads();
        // += relative-position bias (closed form of POS_INDEX) + mask
        {
            float* scf = &u.a.sc[0][0];
            for (int e = tid; e < TT * TT; e += 256) {
                int i = e / TT, j = e % TT;
                int zi = i / 72, hi_ = (i / 12) % 6, wi = i % 12;
                int zj = j / 72, hj  = (j / 12) % 6, wj = j % 12;
                int pos = (zi + 2 * zj) * 828 + (hi_ + 6 * hj) * 23 + (wi - wj + 11);
                scf[e] += btab[(size_t)pos * 60 + w * 6 + h] + ml[e];
            }
        }
        __syncthreads();
        // softmax over j, row per thread; emit bf16 attn with zero K-padding
        if (tid < TT) {
            float mx = -3.4e38f;
            for (int j = 0; j < TT; ++j) mx = fmaxf(mx, u.a.sc[tid][j]);
            float s = 0.f;
            for (int j = 0; j < TT; ++j) {
                float e = __expf(u.a.sc[tid][j] - mx);
                u.a.sc[tid][j] = e; s += e;
            }
            float inv = 1.0f / s;
            for (int j = 0; j < TT; ++j) u.a.at[tid][j] = (__bf16)(u.a.sc[tid][j] * inv);
            for (int j = TT; j < TP; ++j) u.a.at[tid][j] = (__bf16)0.0f;
        }
        __syncthreads();
        // out_h = attn @ v_h -> overwrite qo[h] (q_h is dead); 9x2 tiles, K = 160
        for (int t = wv; t < 18; t += 8) {
            int mt = t / 2, nt = t % 2;
            const __bf16* arow = &u.a.at[mt * 16 + l16][0];
            const __bf16* bcol = &vt[h][nt * 16 + l16][0];
            v8f acc = {};
#pragma unroll
            for (int ks = 0; ks < 5; ++ks) {
                v16bf A = load_a_frag(arow + ks * 32, hi8);
                v16bf B = *reinterpret_cast<const v16bf*>(bcol + ks * 32 + kh16);
                acc = __builtin_amdgcn_wmma_f32_16x16x32_bf16(
                        false, A, false, B, (short)0, acc, false, false);
            }
            int dloc = nt * 16 + l16, mb = mt * 16 + hi8;
#pragma unroll
            for (int r = 0; r < 8; ++r) qo[h][mb + r][dloc] = (__bf16)acc[r];
        }
        __syncthreads();
    }

    // ---- GEMM2: out = attn_out @ W2 + b2 -----------------------------------
    for (int i = tid; i < DD * DD; i += 256) {           // W2^T -> [n][k]
        int n = i % DD, k = i / DD;
        u.g2.w2[n][k] = (__bf16)W2[(size_t)k * DD + n];
    }
    __syncthreads();
    float* og = out + (size_t)lw * TT * DD;
    for (int t = wv; t < 9 * 12; t += 8) {
        int mt = t / 12, nt = t % 12;
        int m = mt * 16 + l16;
        const __bf16* bcol = &u.g2.w2[nt * 16 + l16][0];
        v8f acc = {};
#pragma unroll
        for (int ks = 0; ks < 6; ++ks) {                 // K-step ks lives in head ks
            v16bf A = load_a_frag(&qo[ks][m][0], hi8);
            v16bf B = *reinterpret_cast<const v16bf*>(bcol + ks * 32 + kh16);
            acc = __builtin_amdgcn_wmma_f32_16x16x32_bf16(
                    false, A, false, B, (short)0, acc, false, false);
        }
        int n = nt * 16 + l16;
        float bv = b2s[n];
        int mb = mt * 16 + hi8;
#pragma unroll
        for (int r = 0; r < 8; ++r)
            og[(size_t)(mb + r) * DD + n] = acc[r] + bv;
    }
}

extern "C" void kernel_launch(void* const* d_in, const int* in_sizes, int n_in,
                              void* d_out, int out_size, void* d_ws, size_t ws_size,
                              hipStream_t stream) {
    (void)in_sizes; (void)n_in; (void)d_ws; (void)ws_size; (void)out_size;
    const float* x    = (const float*)d_in[0];
    const float* mask = (const float*)d_in[1];
    const float* W1   = (const float*)d_in[2];
    const float* b1   = (const float*)d_in[3];
    const float* W2   = (const float*)d_in[4];
    const float* b2   = (const float*)d_in[5];
    const float* bt   = (const float*)d_in[6];
    float* out = (float*)d_out;
    pangu_fused<<<dim3(NLON * NWIN), dim3(256), 0, stream>>>(x, mask, W1, b1, W2, b2, bt, out);
}
